// PWCNetMultiScale_1133871366497
// MI455X (gfx1250) — matrix operand order repaired and checked by reading, not compiled, not run on hardware
//
#include <hip/hip_runtime.h>

typedef _Float16 half_t;
typedef __attribute__((ext_vector_type(16))) _Float16 v16h;
typedef __attribute__((ext_vector_type(8)))  _Float16 v8h;
typedef __attribute__((ext_vector_type(8)))  float    v8f;

// ===========================================================================
// Weight packing (NHWC / K = (r*3+s)*Cp + c enumeration) -> WMMA A-fragments.
// Apack layout: ((mt*kSteps + ks)*32 + lane)*16 + e
// A-fragment (16-bit A 16x32), contiguous k order used here (B matches): row M = lane%16;
//   lanes<16 : e0..15 -> K=e
//   lanes>=16: e0..15 -> K=e+16
// ===========================================================================
__global__ void pack_weights_nhwc(const float* __restrict__ W, half_t* __restrict__ Apack,
                                  int Cout, int Cin, int Cp, int mTiles, int kSteps) {
    int idx = blockIdx.x * blockDim.x + threadIdx.x;
    int total = mTiles * kSteps * 32 * 16;
    if (idx >= total) return;
    int e    = idx & 15;
    int lane = (idx >> 4) & 31;
    int rest = idx >> 9;
    int ks   = rest % kSteps;
    int mt   = rest / kSteps;
    int m = mt * 16 + (lane & 15);
    int k = ks * 32 + e + (lane < 16 ? 0 : 16);
    int rs = k / Cp;          // kernel tap index r*3+s
    int c  = k % Cp;          // input channel
    float v = 0.0f;
    if (m < Cout && c < Cin) v = W[((size_t)m * Cin + c) * 9 + rs];
    Apack[idx] = (half_t)v;
}

// Channel-major enumeration (k = c*9 + rs) for the tiny FE conv1 (K=27 pad 32).
__global__ void pack_weights_cfirst(const float* __restrict__ W, half_t* __restrict__ Apack,
                                    int Cout, int K) {
    int idx = blockIdx.x * blockDim.x + threadIdx.x;   // one 16x32 tile: 512 halves
    if (idx >= 512) return;
    int e = idx & 15, lane = idx >> 4;
    int m = lane & 15;
    int k = e + (lane < 16 ? 0 : 16);
    float v = 0.0f;
    if (m < Cout && k < K) v = W[(size_t)m * K + k];
    Apack[idx] = (half_t)v;
}

// ===========================================================================
// Main conv2d: NHWC padded f16 in -> NHWC padded f16 out (+opt NCHW f32 out).
// N-blocked: each wave computes 16 channels x 64 pixels (4 N-tiles), sharing
// one A fragment across 4 B fragments / 4 independent accumulator chains.
// Per K-step: 2 A loads + 8 B loads + 4 WMMAs.
// Input: [B][Hin+2Pin][Win+2Pin][Cp], halo & channel-pad zeroed, Cp%32==0.
// ===========================================================================
__global__ __launch_bounds__(256) void conv_nhwc_wmma(
    const half_t* __restrict__ in, const half_t* __restrict__ Apack,
    const float* __restrict__ bias,
    half_t* __restrict__ outH, float* __restrict__ outF,
    int B, int Cp, int Hin, int Win, int Pin,
    int Cout, int Cpo, int Hout, int Wout, int Pout,
    int stride, int pad, int dil, int relu, int csteps)
{
    const int lane = threadIdx.x & 31;
    const int wave = threadIdx.x >> 5;
    const int ntBase = (blockIdx.x * 8 + wave) * 4;
    const int mt = blockIdx.y;
    const int Ntot = B * Hout * Wout;
    const int col = lane & 15;
    const int Wp = Win + 2 * Pin;
    const int Hp = Hin + 2 * Pin;
    const int kHalf = (lane < 16) ? 0 : 16;

    bool val[4];
    int ohs[4], ows[4], bbs[4];
    const half_t* pix[4];
    #pragma unroll
    for (int j = 0; j < 4; ++j) {
        int n = (ntBase + j) * 16 + col;
        val[j] = (n < Ntot);
        int nc = val[j] ? n : 0;
        int ow = nc % Wout;
        int tt = nc / Wout;
        int oh = tt % Hout;
        int b  = tt / Hout;
        ohs[j] = oh; ows[j] = ow; bbs[j] = b;
        int y0 = oh * stride - pad + Pin;
        int x0 = ow * stride - pad + Pin;
        pix[j] = in + (((size_t)b * Hp + y0) * Wp + x0) * Cp + kHalf;
    }

    v8f acc[4] = {};
    const half_t* ap = Apack + (size_t)mt * (9 * csteps) * 512 + lane * 16;

    #pragma unroll
    for (int r = 0; r < 3; ++r) {
        #pragma unroll
        for (int s = 0; s < 3; ++s) {
            const size_t off = ((size_t)(r * dil) * Wp + s * dil) * Cp;
            for (int cs = 0; cs < csteps; ++cs) {
                v16h a = *(const v16h*)ap;
                __builtin_prefetch(ap + 512, 0, 3);
                ap += 512;
                #pragma unroll
                for (int j = 0; j < 4; ++j) {
                    v16h bf = *(const v16h*)(pix[j] + off + cs * 32);
                    acc[j] = __builtin_amdgcn_wmma_f32_16x16x32_f16(false, a, false, bf,
                                                                    (short)0, acc[j], false, false);
                }
            }
        }
    }

    const int mbase = mt * 16 + (lane >= 16 ? 8 : 0);
    const int Wpo = Wout + 2 * Pout;
    #pragma unroll
    for (int j = 0; j < 4; ++j) {
        v8h hv;
        #pragma unroll
        for (int i = 0; i < 8; ++i) {
            int m = mbase + i;
            float bi = (m < Cout) ? bias[m] : 0.0f;
            float v = acc[j][i] + bi;
            if (relu) v = fmaxf(v, 0.0f);
            if (m >= Cout) v = 0.0f;
            hv[i] = (half_t)v;
        }
        if (val[j]) {
            half_t* op = outH + (((size_t)bbs[j] * (Hout + 2 * Pout) + (ohs[j] + Pout)) * Wpo
                                 + (ows[j] + Pout)) * Cpo + mbase;
            *(v8h*)op = hv;
            if (outF) {
                #pragma unroll
                for (int i = 0; i < 8; ++i) {
                    int m = mbase + i;
                    if (m < Cout) {
                        float v = acc[j][i] + bias[m];
                        if (relu) v = fmaxf(v, 0.0f);
                        outF[(((size_t)bbs[j] * Cout + m) * Hout + ohs[j]) * Wout + ows[j]] = v;
                    }
                }
            }
        }
    }
}

// ===========================================================================
// FE conv1 specialization: float NCHW image (Cin=3, 3x3, stride2, pad1) ->
// NHWC padded f16 (Cout=16, Cpo=32). K=27 padded to 32, single WMMA step,
// gather fully unrolled with compile-time (c,r,s).
// ===========================================================================
__global__ __launch_bounds__(256) void fe_conv1_wmma(
    const float* __restrict__ img, const half_t* __restrict__ Apack,
    const float* __restrict__ bias, half_t* __restrict__ outH,
    int B, int Hin, int Win, int Hout, int Wout, int Pout, int Cpo)
{
    const int lane = threadIdx.x & 31;
    const int wave = threadIdx.x >> 5;
    const int nt = blockIdx.x * 8 + wave;
    const int Ntot = B * Hout * Wout;
    const int col = lane & 15;

    int n = nt * 16 + col;
    bool nvalid = (n < Ntot);
    int nc = nvalid ? n : 0;
    int ow = nc % Wout;
    int t  = nc / Wout;
    int oh = t % Hout;
    int b  = t / Hout;
    int ih0 = oh * 2 - 1;
    int iw0 = ow * 2 - 1;
    const float* ib = img + (size_t)b * 3 * Hin * Win;

    v16h a = *(const v16h*)(Apack + lane * 16);
    v16h bb;
    auto g = [&](int k) -> half_t {           // k is a compile-time constant after unroll
        if (k >= 27) return (half_t)0.0f;
        const int c = k / 9, rs = k % 9, r = rs / 3, s = rs % 3;
        int ih = ih0 + r, iw = iw0 + s;
        float v = 0.0f;
        if ((unsigned)ih < (unsigned)Hin && (unsigned)iw < (unsigned)Win)
            v = ib[((size_t)c * Hin + ih) * Win + iw];
        return (half_t)v;
    };
    if (lane < 16) {
        #pragma unroll
        for (int i = 0; i < 16; ++i) bb[i] = g(i);
    } else {
        #pragma unroll
        for (int i = 0; i < 16; ++i) bb[i] = g(16 + i);
    }

    v8f acc = {};
    acc = __builtin_amdgcn_wmma_f32_16x16x32_f16(false, a, false, bb, (short)0, acc, false, false);

    const int mbase = (lane >= 16) ? 8 : 0;
    v8h hv;
    #pragma unroll
    for (int i = 0; i < 8; ++i) {
        float v = fmaxf(acc[i] + bias[mbase + i], 0.0f);   // relu
        hv[i] = (half_t)v;
    }
    if (nvalid) {
        const int Wpo = Wout + 2 * Pout;
        half_t* op = outH + (((size_t)b * (Hout + 2 * Pout) + (oh + Pout)) * Wpo + (ow + Pout)) * Cpo + mbase;
        *(v8h*)op = hv;
    }
}

// ===========================================================================
// Elementwise / data-movement kernels
// ===========================================================================
// 2x2 average == jax bilinear half-resize (antialias=False), float NCHW.
__global__ void downsample2x_kernel(const float* __restrict__ in, float* __restrict__ out,
                                    int B, int C, int Hout, int Wout) {
    int idx = blockIdx.x * blockDim.x + threadIdx.x;
    int total = B * C * Hout * Wout;
    if (idx >= total) return;
    int w = idx % Wout; int t = idx / Wout;
    int h = t % Hout;   t /= Hout;
    int c = t % C;      int b = t / C;
    int Win = Wout * 2;
    const float* p = in + (((size_t)b * C + c) * (Hout * 2) + 2 * h) * Win + 2 * w;
    out[idx] = 0.25f * (p[0] + p[1] + p[Win] + p[Win + 1]);
}

// resize_double(flow) * 2.0, float NCHW, edge clamp.
__global__ void upsample2x_flow_kernel(const float* __restrict__ in, float* __restrict__ out,
                                       int B, int Hin, int Win) {
    int Hout = Hin * 2, Wout = Win * 2;
    int idx = blockIdx.x * blockDim.x + threadIdx.x;
    int total = B * 2 * Hout * Wout;
    if (idx >= total) return;
    int w = idx % Wout; int t = idx / Wout;
    int h = t % Hout;   t /= Hout;
    int c = t % 2;      int b = t / 2;
    float sy = fminf(fmaxf((h + 0.5f) * 0.5f - 0.5f, 0.0f), (float)(Hin - 1));
    float sx = fminf(fmaxf((w + 0.5f) * 0.5f - 0.5f, 0.0f), (float)(Win - 1));
    float y0f = floorf(sy), x0f = floorf(sx);
    int y0 = (int)y0f, x0 = (int)x0f;
    int y1 = min(y0 + 1, Hin - 1), x1 = min(x0 + 1, Win - 1);
    float wy = sy - y0f, wx = sx - x0f;
    const float* p = in + ((size_t)b * 2 + c) * Hin * Win;
    float v = p[y0 * Win + x0] * (1 - wx) * (1 - wy) + p[y0 * Win + x1] * wx * (1 - wy)
            + p[y1 * Win + x0] * (1 - wx) * wy       + p[y1 * Win + x1] * wx * wy;
    out[idx] = 2.0f * v;
}

// grid_sample_border with PWC-Net flow_to_grid convention. f2/out: NHWC halo4, C=64.
__global__ void warp_kernel(const half_t* __restrict__ f2, const float* __restrict__ flow,
                            half_t* __restrict__ out, int B, int H, int W) {
    int idx = blockIdx.x * blockDim.x + threadIdx.x;
    int total = B * H * W;
    if (idx >= total) return;
    int w = idx % W; int t = idx / W;
    int h = t % H;   int b = t / H;
    size_t fb = ((size_t)b * 2) * H * W + (size_t)h * W + w;
    float f0 = flow[fb];
    float f1v = flow[fb + (size_t)H * W];
    float gx = ((float)h + f0) * 2.0f / (float)(W - 1) - 1.0f;
    float gy = ((float)w + f1v) * 2.0f / (float)(H - 1) - 1.0f;
    float x = fminf(fmaxf((gx + 1.0f) * W * 0.5f - 0.5f, 0.0f), (float)(W - 1));
    float y = fminf(fmaxf((gy + 1.0f) * H * 0.5f - 0.5f, 0.0f), (float)(H - 1));
    float x0f = floorf(x), y0f = floorf(y);
    int x0 = (int)x0f, y0 = (int)y0f;
    int x1 = min(x0 + 1, W - 1), y1 = min(y0 + 1, H - 1);
    float wx = x - x0f, wy = y - y0f;
    float w00 = (1 - wx) * (1 - wy), w01 = wx * (1 - wy);
    float w10 = (1 - wx) * wy,       w11 = wx * wy;
    const int Wp = W + 8;
    const half_t* base = f2 + (size_t)b * (H + 8) * Wp * 64;
    const half_t* p00 = base + ((size_t)(y0 + 4) * Wp + (x0 + 4)) * 64;
    const half_t* p01 = base + ((size_t)(y0 + 4) * Wp + (x1 + 4)) * 64;
    const half_t* p10 = base + ((size_t)(y1 + 4) * Wp + (x0 + 4)) * 64;
    const half_t* p11 = base + ((size_t)(y1 + 4) * Wp + (x1 + 4)) * 64;
    half_t* ob = out + ((size_t)b * (H + 8) + (h + 4)) * Wp * 64 + (size_t)(w + 4) * 64;
    for (int c = 0; c < 64; ++c) {
        float v = (float)p00[c] * w00 + (float)p01[c] * w01
                + (float)p10[c] * w10 + (float)p11[c] * w11;
        ob[c] = (half_t)v;
    }
}

// copy f1 (NHWC halo4,C64) -> estIn channels [0,64) (NHWC halo1,Cp160)
__global__ void concat_f1_kernel(const half_t* __restrict__ f1, half_t* __restrict__ estIn,
                                 int B, int H, int W) {
    int idx = blockIdx.x * blockDim.x + threadIdx.x;
    int total = B * H * W * 64;
    if (idx >= total) return;
    int c = idx & 63; int t = idx >> 6;
    int w = t % W;    t /= W;
    int h = t % H;    int b = t / H;
    half_t v = f1[(((size_t)b * (H + 8) + (h + 4)) * (W + 8) + (w + 4)) * 64 + c];
    estIn[(((size_t)b * (H + 2) + (h + 1)) * (W + 2) + (w + 1)) * 160 + c] = v;
}

// cost volume -> estIn channels [64,145). f1/f2w NHWC halo4 (zeros give free pad).
__global__ void costvol_kernel(const half_t* __restrict__ f1, const half_t* __restrict__ f2w,
                               half_t* __restrict__ estIn, int B, int H, int W) {
    int idx = blockIdx.x * blockDim.x + threadIdx.x;
    int total = B * H * W * 81;
    if (idx >= total) return;
    int d = idx % 81; int t = idx / 81;
    int w = t % W;    t /= W;
    int h = t % H;    int b = t / H;
    int di = d / 9 - 4;
    int dj = d % 9 - 4;
    const int Wp = W + 8;
    const half_t* a = f1  + (((size_t)b * (H + 8) + (h + 4))      * Wp + (w + 4))      * 64;
    const half_t* p = f2w + (((size_t)b * (H + 8) + (h + 4 - di)) * Wp + (w + 4 - dj)) * 64;
    float s = 0.0f;
    for (int c = 0; c < 64; ++c) s += (float)a[c] * (float)p[c];
    estIn[(((size_t)b * (H + 2) + (h + 1)) * (W + 2) + (w + 1)) * 160 + 64 + d] = (half_t)(s * (1.0f / 64.0f));
}

// ===========================================================================
// Host orchestration
// ===========================================================================
struct Layer { const float* w; const float* b; };
static inline int cdiv(int a, int b) { return (a + b - 1) / b; }
static inline int cp32(int c) { return ((c + 31) / 32) * 32; }

extern "C" void kernel_launch(void* const* d_in, const int* in_sizes, int n_in,
                              void* d_out, int out_size, void* d_ws, size_t ws_size,
                              hipStream_t stream) {
    const int B = 8;
    const float* img1 = (const float*)d_in[0];
    const float* img2 = (const float*)d_in[1];

    // ---- resolve parameter flattening order ----
    Layer fe[3], est[3][4], ctx[7];
    int i2 = in_sizes[2];
    bool ctxFirst = (i2 == 128 || i2 == 2304);
    bool bFirst   = ctxFirst ? (i2 == 128) : (i2 == 16);
    int pi = 2;
    auto take = [&](Layer& L) {
        if (bFirst) { L.b = (const float*)d_in[pi++]; L.w = (const float*)d_in[pi++]; }
        else        { L.w = (const float*)d_in[pi++]; L.b = (const float*)d_in[pi++]; }
    };
    if (ctxFirst) {
        for (int i = 0; i < 7; ++i) take(ctx[i]);
        for (int l = 0; l < 3; ++l) for (int i = 0; i < 4; ++i) take(est[l][i]);
        for (int i = 0; i < 3; ++i) take(fe[i]);
    } else {
        for (int i = 0; i < 3; ++i) take(fe[i]);
        for (int l = 0; l < 3; ++l) for (int i = 0; i < 4; ++i) take(est[l][i]);
        for (int i = 0; i < 7; ++i) take(ctx[i]);
    }

    // ---- bump allocator ----
    char* wsBase = (char*)d_ws;
    size_t wsOff = 0;
    auto balloc = [&](size_t bytes) -> void* {
        size_t a = (wsOff + 255) & ~(size_t)255;
        wsOff = a + bytes;
        return wsBase + a;
    };
    auto clr = [&](void* p, size_t bytes) { hipMemsetAsync(p, 0, bytes, stream); };

    // float NCHW pyramids
    float* p1l1 = (float*)balloc((size_t)B * 3 * 256 * 256 * 4);
    float* p1l2 = (float*)balloc((size_t)B * 3 * 128 * 128 * 4);
    float* p2l1 = (float*)balloc((size_t)B * 3 * 256 * 256 * 4);
    float* p2l2 = (float*)balloc((size_t)B * 3 * 128 * 128 * 4);

    // NHWC padded f16 activations (sized for the largest level)
    const size_t t1B    = (size_t)B * 258 * 258 * 32 * 2;   // FE conv1 out, halo1, Cp32
    const size_t t2B    = (size_t)B * 130 * 130 * 32 * 2;   // FE conv2 out, halo1, Cp32
    const size_t fB     = (size_t)B * 72  * 72  * 64 * 2;   // features, halo4, Cp64
    const size_t estInB = (size_t)B * 66  * 66  * 160 * 2;  // concat input, halo1, Cp160
    const size_t x1B    = (size_t)B * 66  * 66  * 128 * 2;
    const size_t x2B    = (size_t)B * 66  * 66  * 64 * 2;
    const size_t x3B    = (size_t)B * 66  * 66  * 32 * 2;
    const size_t ctxBB  = (size_t)B * 96  * 96  * 128 * 2;  // ctx ping/pong, halo16
    const size_t flowHB = (size_t)B * 96  * 96  * 32 * 2;   // flow f16, halo16, Cp32
    half_t* t1    = (half_t*)balloc(t1B);
    half_t* t2    = (half_t*)balloc(t2B);
    half_t* f1b   = (half_t*)balloc(fB);
    half_t* f2b   = (half_t*)balloc(fB);
    half_t* f2wb  = (half_t*)balloc(fB);
    half_t* estIn = (half_t*)balloc(estInB);
    half_t* x1    = (half_t*)balloc(x1B);
    half_t* x2    = (half_t*)balloc(x2B);
    half_t* x3    = (half_t*)balloc(x3B);
    half_t* ctxA  = (half_t*)balloc(ctxBB);
    half_t* ctxBp = (half_t*)balloc(ctxBB);
    half_t* flowH = (half_t*)balloc(flowHB);
    float*  flow0F = (float*)balloc((size_t)B * 2 * 64 * 64 * 4);
    float*  flowUp = (float*)balloc((size_t)B * 2 * 64 * 64 * 4);

    // ---- pack weights ----
    auto packN = [&](const Layer& L, int cin, int cout) -> half_t* {
        int Cp = cp32(cin), kSteps = 9 * Cp / 32, mTiles = cdiv(cout, 16);
        size_t nH = (size_t)mTiles * kSteps * 512;
        half_t* p = (half_t*)balloc(nH * 2);
        pack_weights_nhwc<<<cdiv((int)nH, 256), 256, 0, stream>>>(L.w, p, cout, cin, Cp, mTiles, kSteps);
        return p;
    };
    half_t* feP0 = (half_t*)balloc(512 * 2);
    pack_weights_cfirst<<<2, 256, 0, stream>>>(fe[0].w, feP0, 16, 27);
    half_t* feP1 = packN(fe[1], 16, 32);
    half_t* feP2 = packN(fe[2], 32, 64);
    const int estDims[4][2] = {{145,128},{128,64},{64,32},{32,2}};
    const int ctxDims[7][2] = {{2,128},{128,128},{128,128},{128,96},{96,64},{64,32},{32,2}};
    half_t* estP[3][4]; half_t* ctxP[7];
    for (int l = 0; l < 3; ++l)
        for (int i = 0; i < 4; ++i) estP[l][i] = packN(est[l][i], estDims[i][0], estDims[i][1]);
    for (int i = 0; i < 7; ++i) ctxP[i] = packN(ctx[i], ctxDims[i][0], ctxDims[i][1]);

    auto convN = [&](const half_t* in, const half_t* Ap, const float* bias,
                     half_t* outH, float* outF,
                     int Cin, int Hin, int Win, int Pin,
                     int Cout, int Cpo, int Hout, int Wout, int Pout,
                     int stride, int pad, int dil, int relu) {
        int Cp = cp32(Cin), csteps = Cp / 32, mTiles = cdiv(Cout, 16);
        int Ntot = B * Hout * Wout;
        int waves = cdiv(Ntot, 64);                 // 4 N-tiles (64 pixels) per wave
        dim3 grid(cdiv(waves, 8), mTiles);
        conv_nhwc_wmma<<<grid, 256, 0, stream>>>(in, Ap, bias, outH, outF,
            B, Cp, Hin, Win, Pin, Cout, Cpo, Hout, Wout, Pout, stride, pad, dil, relu, csteps);
    };

    // ---- pyramids ----
    downsample2x_kernel<<<cdiv(B*3*256*256,256),256,0,stream>>>(img1, p1l1, B, 3, 256, 256);
    downsample2x_kernel<<<cdiv(B*3*128*128,256),256,0,stream>>>(p1l1, p1l2, B, 3, 128, 128);
    downsample2x_kernel<<<cdiv(B*3*256*256,256),256,0,stream>>>(img2, p2l1, B, 3, 256, 256);
    downsample2x_kernel<<<cdiv(B*3*128*128,256),256,0,stream>>>(p2l1, p2l2, B, 3, 128, 128);

    const float* pyr1[3] = {img1, p1l1, p1l2};
    const float* pyr2[3] = {img2, p2l1, p2l2};
    const int Sl[3] = {512, 256, 128};

    auto runFE = [&](const float* img, int S, half_t* fout) {
        int H1 = S / 2, H2 = S / 4, H3 = S / 8;
        clr(t1, t1B);
        {
            int Ntot = B * H1 * H1;
            dim3 grid(cdiv(cdiv(Ntot, 16), 8), 1);
            fe_conv1_wmma<<<grid, 256, 0, stream>>>(img, feP0, fe[0].b, t1, B, S, S, H1, H1, 1, 32);
        }
        clr(t2, t2B);
        convN(t1, feP1, fe[1].b, t2, nullptr, 16, H1, H1, 1, 32, 32, H2, H2, 1, 2, 1, 1, 1);
        clr(fout, fB);
        convN(t2, feP2, fe[2].b, fout, nullptr, 32, H2, H2, 1, 64, 64, H3, H3, 4, 2, 1, 1, 1);
    };

    float* outFlow2 = (float*)d_out;                           // [8,2,16,16]
    float* outFlow1 = (float*)d_out + 8*2*16*16;               // [8,2,32,32]
    float* outCtx   = (float*)d_out + 8*2*16*16 + 8*2*32*32;   // [8,2,64,64]

    float* flowPrev = nullptr;
    int Hprev = 0;
    for (int level = 2; level >= 0; --level) {
        int S = Sl[level], H = S / 8, W = H;
        runFE(pyr1[level], S, f1b);
        runFE(pyr2[level], S, f2b);

        const half_t* f2w = f2b;
        if (flowPrev) {
            upsample2x_flow_kernel<<<cdiv(B*2*H*W,256),256,0,stream>>>(flowPrev, flowUp, B, Hprev, Hprev);
            clr(f2wb, fB);
            warp_kernel<<<cdiv(B*H*W,256),256,0,stream>>>(f2b, flowUp, f2wb, B, H, W);
            f2w = f2wb;
        }

        clr(estIn, estInB);
        concat_f1_kernel<<<cdiv(B*H*W*64,256),256,0,stream>>>(f1b, estIn, B, H, W);
        costvol_kernel<<<cdiv(B*H*W*81,256),256,0,stream>>>(f1b, f2w, estIn, B, H, W);

        float* flowOutF = (level == 2) ? outFlow2 : (level == 1) ? outFlow1 : flow0F;
        clr(x1, x1B);
        convN(estIn, estP[level][0], est[level][0].b, x1, nullptr, 145, H, W, 1, 128, 128, H, W, 1, 1, 1, 1, 1);
        clr(x2, x2B);
        convN(x1, estP[level][1], est[level][1].b, x2, nullptr, 128, H, W, 1, 64, 64, H, W, 1, 1, 1, 1, 1);
        clr(x3, x3B);
        convN(x2, estP[level][2], est[level][2].b, x3, nullptr, 64, H, W, 1, 32, 32, H, W, 1, 1, 1, 1, 1);
        clr(flowH, flowHB);
        convN(x3, estP[level][3], est[level][3].b, flowH, flowOutF, 32, H, W, 1, 2, 32, H, W, 16, 1, 1, 1, 0);

        flowPrev = flowOutF;
        Hprev = H;
    }

    // ---- context network on level-0 flow (flowH: NHWC halo16 Cp32) ----
    {
        int H = 64, W = 64;
        clr(ctxA, ctxBB);
        convN(flowH, ctxP[0], ctx[0].b, ctxA, nullptr, 2,   H, W, 16, 128, 128, H, W, 16, 1, 1,  1,  1);
        clr(ctxBp, ctxBB);
        convN(ctxA, ctxP[1], ctx[1].b, ctxBp, nullptr, 128, H, W, 16, 128, 128, H, W, 16, 1, 2,  2,  1);
        clr(ctxA, ctxBB);
        convN(ctxBp, ctxP[2], ctx[2].b, ctxA, nullptr, 128, H, W, 16, 128, 128, H, W, 16, 1, 4,  4,  1);
        clr(ctxBp, ctxBB);
        convN(ctxA, ctxP[3], ctx[3].b, ctxBp, nullptr, 128, H, W, 16, 96,  96,  H, W, 16, 1, 8,  8,  1);
        clr(ctxA, ctxBB);
        convN(ctxBp, ctxP[4], ctx[4].b, ctxA, nullptr, 96,  H, W, 16, 64,  64,  H, W, 16, 1, 16, 16, 1);
        clr(ctxBp, ctxBB);
        convN(ctxA, ctxP[5], ctx[5].b, ctxBp, nullptr, 64,  H, W, 16, 32,  32,  H, W, 16, 1, 1,  1,  1);
        clr(flowH, flowHB);
        convN(ctxBp, ctxP[6], ctx[6].b, flowH, outCtx, 32,  H, W, 16, 2,   32,  H, W, 16, 1, 1,  1,  0);
    }

    (void)n_in; (void)out_size; (void)ws_size;
}
